// GM_GCN_42726334660716
// MI455X (gfx1250) — compile-verified, matrix operand order checked by reading, not run on hardware
//
#include <hip/hip_runtime.h>

#define N_NODES 50000
#define N_EDGES 1600000
#define DIM 128
#define NCLS 40

typedef __attribute__((ext_vector_type(16))) __bf16 v16bf;
typedef __attribute__((ext_vector_type(8))) float v8f;

// f32 -> bf16 round-to-nearest-even
static __device__ __forceinline__ unsigned short f2bf(float f) {
  unsigned int u = __float_as_uint(f);
  unsigned int r = (u + 0x7FFFu + ((u >> 16) & 1u)) >> 16;
  return (unsigned short)r;
}

// ---------------- degree / norm ----------------
__global__ void k_deg_init(float* __restrict__ deg, int n) {
  int i = blockIdx.x * blockDim.x + threadIdx.x;
  if (i < n) deg[i] = 1.0f;  // self-loop contribution
}

__global__ void k_deg_count(const int* __restrict__ ei, float* __restrict__ deg, int e) {
  int i = blockIdx.x * blockDim.x + threadIdx.x;
  if (i < e) atomicAdd(&deg[ei[N_EDGES + i]], 1.0f);  // in-degree of targets (col)
}

__global__ void k_dinv(float* __restrict__ deg, int n) {
  int i = blockIdx.x * blockDim.x + threadIdx.x;
  if (i < n) deg[i] = rsqrtf(deg[i]);  // deg >= 1 always
}

// ---------------- f32 -> bf16 feature conversion ----------------
__global__ void k_cvt(const float* __restrict__ x, unsigned short* __restrict__ out, int n4) {
  int i = blockIdx.x * blockDim.x + threadIdx.x;
  if (i >= n4) return;
  float4 v = ((const float4*)x)[i];
  unsigned int lo = (unsigned int)f2bf(v.x) | ((unsigned int)f2bf(v.y) << 16);
  unsigned int hi = (unsigned int)f2bf(v.z) | ((unsigned int)f2bf(v.w) << 16);
  ((uint2*)out)[i] = make_uint2(lo, hi);
}

// ---------------- pack W into wave32 WMMA B-operand layout ----------------
// tile (nt,kt): 32 lanes x 16 bf16 each. lane: n = nt*16 + (lane&15);
// element j: k = kt*32 + ((j<8) ? (hi*8+j) : (16 + hi*8 + j-8)), hi = lane>>4.
__global__ void k_pack_b(const float* __restrict__ W, unsigned short* __restrict__ Bp,
                         int ntiles, int ncols_src, int transpose) {
  int idx = blockIdx.x * blockDim.x + threadIdx.x;
  if (idx >= ntiles * 2048) return;
  int j    = idx & 15;
  int lane = (idx >> 4) & 31;
  int kt   = (idx >> 9) & 3;
  int nt   = idx >> 11;
  int l = lane & 15, hi = lane >> 4;
  int n = nt * 16 + l;
  int kk = (j < 8) ? (hi * 8 + j) : (16 + hi * 8 + (j - 8));
  int k = kt * 32 + kk;
  float v = 0.0f;
  if (transpose) {                      // B[k][n] = Wsrc[n][k]  (Wout: NCLS x DIM)
    if (n < ncols_src) v = W[n * DIM + k];
  } else {                              // B[k][n] = Wsrc[k][n]  (W1/W2: DIM x DIM)
    v = W[k * DIM + n];
  }
  Bp[idx] = f2bf(v);
}

// ---------------- WMMA GEMM: C[M x ncols] = A[M x 128] * B ----------------
// FUSE:  also writes agg[m][n] = dinv[m]^2 * C[m][n]  (self-loop init of scatter accum)
// FINAL: adds bias[n] in the epilogue
template <int NT, int LDC, bool FINAL, bool FUSE>
__global__ __launch_bounds__(256) void k_gemm(const unsigned short* __restrict__ A,
                                              const unsigned short* __restrict__ Bp,
                                              float* __restrict__ C,
                                              float* __restrict__ agg,
                                              const float* __restrict__ extra, // bias or dinv
                                              int M, int ncols) {
  __shared__ __align__(16) unsigned short Bs[NT * 2048];
  {
    const uint4* src = (const uint4*)Bp;
    uint4* dst = (uint4*)Bs;
    for (int i = threadIdx.x; i < NT * 256; i += 256) dst[i] = src[i];
  }
  __syncthreads();

  int wave = threadIdx.x >> 5;
  int lane = threadIdx.x & 31;
  int m0 = blockIdx.x * 128 + wave * 16;
  if (m0 >= M) return;                  // wave-uniform: EXEC stays all-1 for WMMA
  int l = lane & 15, hi = lane >> 4;

  union AV { uint4 q[2]; v16bf v; };
  AV a[4];
  const unsigned short* arow = A + (size_t)(m0 + l) * DIM + hi * 8;
#pragma unroll
  for (int kt = 0; kt < 4; ++kt) {      // two 16B chunks: K = hi*8+[0..7], 16+hi*8+[0..7]
    a[kt].q[0] = *(const uint4*)(arow + kt * 32);
    a[kt].q[1] = *(const uint4*)(arow + kt * 32 + 16);
  }

  float w[8];                           // per-row self-loop weights (hoisted out of nt loop)
  if constexpr (FUSE) {
#pragma unroll
    for (int r = 0; r < 8; ++r) {
      float d = extra[m0 + hi * 8 + r];
      w[r] = d * d;
    }
  }

#pragma unroll
  for (int nt = 0; nt < NT; ++nt) {
    v8f c = {0.f, 0.f, 0.f, 0.f, 0.f, 0.f, 0.f, 0.f};
#pragma unroll
    for (int kt = 0; kt < 4; ++kt) {
      const v16bf* bp = (const v16bf*)(&Bs[(nt * 4 + kt) * 512 + lane * 16]);
      c = __builtin_amdgcn_wmma_f32_16x16x32_bf16(false, a[kt].v, false, *bp,
                                                  (short)0, c, false, false);
    }
    int n = nt * 16 + l;
    if (n < ncols) {
      float* crow = C + (size_t)(m0 + hi * 8) * LDC + n;   // c[r] -> row m0+hi*8+r
      if constexpr (FINAL) {
        float bv = extra[n];
#pragma unroll
        for (int r = 0; r < 8; ++r) crow[r * LDC] = c[r] + bv;
      } else if constexpr (FUSE) {
        float* arw = agg + (size_t)(m0 + hi * 8) * LDC + n;
#pragma unroll
        for (int r = 0; r < 8; ++r) {
          float v = c[r];
          crow[r * LDC] = v;
          arw[r * LDC] = w[r] * v;
        }
      } else {
#pragma unroll
        for (int r = 0; r < 8; ++r) crow[r * LDC] = c[r];
      }
    }
  }
}

// ---------------- aggregation ----------------
// one edge per wave32: gather full 512B row of h, 4 f32 atomics per lane into agg[col]
__global__ __launch_bounds__(256) void k_agg_edges(const int* __restrict__ ei,
                                                   const float* __restrict__ dinv,
                                                   const float* __restrict__ h,
                                                   float* __restrict__ agg) {
  int wave = threadIdx.x >> 5, lane = threadIdx.x & 31;
  int e = blockIdx.x * 8 + wave;
  if (e >= N_EDGES) return;
  int r = ei[e];
  int c = ei[N_EDGES + e];
  float w = dinv[r] * dinv[c];
  float4 hv = *(const float4*)(h + (size_t)r * DIM + lane * 4);
  float* dst = agg + (size_t)c * DIM + lane * 4;
  atomicAdd(dst + 0, w * hv.x);
  atomicAdd(dst + 1, w * hv.y);
  atomicAdd(dst + 2, w * hv.z);
  atomicAdd(dst + 3, w * hv.w);
}

// fused bias + ReLU + f32->bf16 for next layer's GEMM input
__global__ void k_bias_relu_cvt(const float* __restrict__ agg, const float* __restrict__ bias,
                                unsigned short* __restrict__ out, int n4) {
  int i = blockIdx.x * blockDim.x + threadIdx.x;
  if (i >= n4) return;
  float4 v = ((const float4*)agg)[i];
  int d = (i << 2) & 127;
  v.x = fmaxf(v.x + bias[d + 0], 0.0f);
  v.y = fmaxf(v.y + bias[d + 1], 0.0f);
  v.z = fmaxf(v.z + bias[d + 2], 0.0f);
  v.w = fmaxf(v.w + bias[d + 3], 0.0f);
  unsigned int lo = (unsigned int)f2bf(v.x) | ((unsigned int)f2bf(v.y) << 16);
  unsigned int hi = (unsigned int)f2bf(v.z) | ((unsigned int)f2bf(v.w) << 16);
  ((uint2*)out)[i] = make_uint2(lo, hi);
}

extern "C" void kernel_launch(void* const* d_in, const int* in_sizes, int n_in,
                              void* d_out, int out_size, void* d_ws, size_t ws_size,
                              hipStream_t stream) {
  const float* x    = (const float*)d_in[0];
  const int*   ei   = (const int*)d_in[1];   // [2, E]: row = ei[0..E), col = ei[E..2E)
  const float* W1   = (const float*)d_in[2];
  const float* b1   = (const float*)d_in[3];
  const float* W2   = (const float*)d_in[4];
  const float* b2   = (const float*)d_in[5];
  const float* Wout = (const float*)d_in[6];
  const float* bout = (const float*)d_in[7];
  float* out = (float*)d_out;

  char* p = (char*)d_ws;
  auto take = [&](size_t bytes) { char* q = p; p += (bytes + 255) & ~(size_t)255; return q; };
  float*          dinv = (float*)take((size_t)N_NODES * 4);
  unsigned short* bbuf = (unsigned short*)take((size_t)N_NODES * DIM * 2);  // bf16 features
  float*          h    = (float*)take((size_t)N_NODES * DIM * 4);           // GEMM out
  float*          agg  = (float*)take((size_t)N_NODES * DIM * 4);           // scatter accum
  unsigned short* B1   = (unsigned short*)take(8 * 2048 * 2);
  unsigned short* B2   = (unsigned short*)take(8 * 2048 * 2);
  unsigned short* B3   = (unsigned short*)take(3 * 2048 * 2);

  const int T = 256;
  const int n4 = N_NODES * (DIM / 4);
  const int gN  = (N_NODES + T - 1) / T;
  const int gE  = (N_EDGES + T - 1) / T;
  const int g4  = (n4 + T - 1) / T;
  const int gG  = (N_NODES + 127) / 128;   // 8 waves x 16 rows per block
  const int gEW = (N_EDGES + 7) / 8;       // 1 edge per wave, 8 waves per block

  // normalization
  k_deg_init<<<gN, T, 0, stream>>>(dinv, N_NODES);
  k_deg_count<<<gE, T, 0, stream>>>(ei, dinv, N_EDGES);
  k_dinv<<<gN, T, 0, stream>>>(dinv, N_NODES);

  // prep: bf16 features + packed weights
  k_cvt<<<g4, T, 0, stream>>>(x, bbuf, n4);
  k_pack_b<<<(8 * 2048 + T - 1) / T, T, 0, stream>>>(W1, B1, 8, DIM, 0);
  k_pack_b<<<(8 * 2048 + T - 1) / T, T, 0, stream>>>(W2, B2, 8, DIM, 0);
  k_pack_b<<<(3 * 2048 + T - 1) / T, T, 0, stream>>>(Wout, B3, 3, NCLS, 1);

  // layer 1: h = x@W1 (fused: agg = dinv^2*h) ; agg += edges ; relu(agg + b1) -> bf16
  k_gemm<8, DIM, false, true><<<gG, T, 0, stream>>>(bbuf, B1, h, agg, dinv, N_NODES, DIM);
  k_agg_edges<<<gEW, T, 0, stream>>>(ei, dinv, h, agg);
  k_bias_relu_cvt<<<g4, T, 0, stream>>>(agg, b1, bbuf, n4);

  // layer 2
  k_gemm<8, DIM, false, true><<<gG, T, 0, stream>>>(bbuf, B2, h, agg, dinv, N_NODES, DIM);
  k_agg_edges<<<gEW, T, 0, stream>>>(ei, dinv, h, agg);
  k_bias_relu_cvt<<<g4, T, 0, stream>>>(agg, b2, bbuf, n4);

  // classifier: out = h2 @ Wout.T + bout   (ncols = 40, padded to 48)
  k_gemm<3, NCLS, true, false><<<gG, T, 0, stream>>>(bbuf, B3, out, nullptr, bout, N_NODES, NCLS);
}
